// VGG_SMALL_1W1A_CBAP_75153337745961
// MI455X (gfx1250) — 8-bit matrix operand order repaired and checked by reading, compiled, not run on hardware
//
#include <hip/hip_runtime.h>
#include <hip/hip_bf16.h>
#include <cstdint>
#include <cstddef>

typedef int v8i __attribute__((ext_vector_type(8)));

static constexpr size_t OFF_Y  = 0;
static constexpr size_t OFF_B  = 134217728;
static constexpr size_t OFF_WP = OFF_B  + 33554432;
static constexpr size_t OFF_S1 = OFF_WP + 2359296;
static constexpr size_t OFF_S2 = OFF_S1 + 2048;
static constexpr size_t OFF_SC = OFF_S2 + 2048;
static constexpr size_t OFF_SH = OFF_SC + 2048;
static constexpr size_t OFF_P5 = OFF_SH + 2048;

__global__ void conv0_kernel(const float* __restrict__ x, const float* __restrict__ w,
                             float* __restrict__ out, int N, int H, int W, int Ci, int Co) {
    int idx = blockIdx.x * blockDim.x + threadIdx.x;
    int total = N * H * W * Co;
    if (idx >= total) return;
    int o  = idx % Co;  int t = idx / Co;
    int xx = t % W;     t /= W;
    int yy = t % H;     int n = t / H;
    float acc = 0.f;
    for (int ci = 0; ci < Ci; ++ci) {
        for (int dy = 0; dy < 3; ++dy) {
            int iy = yy + dy - 1;
            if ((unsigned)iy >= (unsigned)H) continue;
            for (int dx = 0; dx < 3; ++dx) {
                int ix = xx + dx - 1;
                if ((unsigned)ix >= (unsigned)W) continue;
                acc += x[(((size_t)n * Ci + ci) * H + iy) * W + ix]
                     * w[((o * Ci + ci) * 3 + dy) * 3 + dx];
            }
        }
    }
    out[idx] = acc;
}

__global__ void packw_kernel(const float* __restrict__ w, signed char* __restrict__ wp,
                             int O, int Ci) {
    int idx = blockIdx.x * blockDim.x + threadIdx.x;
    int K = 9 * Ci;
    if (idx >= O * K) return;
    int ci = idx % Ci;
    int p  = (idx / Ci) % 9;
    int o  = idx / K;
    float v = w[(size_t)(o * Ci + ci) * 9 + p];
    wp[(size_t)o * K + p * Ci + ci] = (v > 0.f) ? 1 : ((v < 0.f) ? -1 : 0);
}

__global__ void binconv_wmma_kernel(const signed char* __restrict__ in,
                                    const signed char* __restrict__ wp,
                                    float* __restrict__ out,
                                    int N, int H, int W, int C, int O) {
    extern __shared__ signed char lds_w[];
    const int K = 9 * C;
    const int obase = blockIdx.y * 32;

    {
        const signed char* wsrc = wp + (size_t)obase * K;
        const int total = 32 * K;
        for (int off = threadIdx.x * 16; off < total; off += blockDim.x * 16) {
            unsigned           lds_addr = (unsigned)(unsigned long long)(const void*)(lds_w + off);
            unsigned long long gaddr    = (unsigned long long)(const void*)(wsrc + off);
            asm volatile("global_load_async_to_lds_b128 %0, %1, off"
                         :: "v"(lds_addr), "v"(gaddr) : "memory");
        }
        asm volatile("s_wait_asynccnt 0x0" ::: "memory");
    }
    __syncthreads();

    const int lane  = threadIdx.x & 31;
    const int wave  = threadIdx.x >> 5;
    const int row   = lane & 15;
    const int khalf = lane >> 4;
    const int mbase = (blockIdx.x * (blockDim.x >> 5) + wave) * 32;

    const int HW = H * W;
    const int m0 = mbase + row;
    const int n0 = m0 / HW, q0 = m0 % HW, y0 = q0 / W, x0 = q0 % W;
    const int m1 = m0 + 16;
    const int n1 = m1 / HW, q1 = m1 % HW, y1 = q1 / W, x1 = q1 % W;

    v8i acc00 = {0,0,0,0,0,0,0,0};
    v8i acc01 = {0,0,0,0,0,0,0,0};
    v8i acc10 = {0,0,0,0,0,0,0,0};
    v8i acc11 = {0,0,0,0,0,0,0,0};

    for (int p = 0; p < 9; ++p) {
        const int dy = p / 3, dx = p - dy * 3;
        const int iy0 = y0 + dy - 1, ix0 = x0 + dx - 1;
        const int iy1 = y1 + dy - 1, ix1 = x1 + dx - 1;
        const bool ok0 = (unsigned)iy0 < (unsigned)H && (unsigned)ix0 < (unsigned)W;
        const bool ok1 = (unsigned)iy1 < (unsigned)H && (unsigned)ix1 < (unsigned)W;
        const signed char* pa0 = in + (((size_t)n0 * H + iy0) * W + ix0) * C + khalf * 16;
        const signed char* pa1 = in + (((size_t)n1 * H + iy1) * W + ix1) * C + khalf * 16;
        const int wbase = p * C + khalf * 16;

        for (int c0 = 0; c0 < C; c0 += 64) {
            v8i a0, a1;
            #pragma unroll
            for (int i = 0; i < 4; ++i) {
                int2 v = ok0 ? *(const int2*)(pa0 + c0 + (i & 1) * 8 + (i >> 1) * 32) : make_int2(0, 0);
                a0[2 * i] = v.x;  a0[2 * i + 1] = v.y;
                int2 u = ok1 ? *(const int2*)(pa1 + c0 + (i & 1) * 8 + (i >> 1) * 32) : make_int2(0, 0);
                a1[2 * i] = u.x;  a1[2 * i + 1] = u.y;
            }
            v8i b0, b1;
            {
                const int base0 = row * K + wbase + c0;
                int4 lo = *(const int4*)(lds_w + base0);
                int4 hi = *(const int4*)(lds_w + base0 + 32);
                b0[0] = lo.x; b0[1] = lo.y; b0[2] = lo.z; b0[3] = lo.w;
                b0[4] = hi.x; b0[5] = hi.y; b0[6] = hi.z; b0[7] = hi.w;
                const int base1 = (row + 16) * K + wbase + c0;
                int4 lo1 = *(const int4*)(lds_w + base1);
                int4 hi1 = *(const int4*)(lds_w + base1 + 32);
                b1[0] = lo1.x; b1[1] = lo1.y; b1[2] = lo1.z; b1[3] = lo1.w;
                b1[4] = hi1.x; b1[5] = hi1.y; b1[6] = hi1.z; b1[7] = hi1.w;
            }
            acc00 = __builtin_amdgcn_wmma_i32_16x16x64_iu8(true, a0, true, b0, acc00, false, false);
            acc01 = __builtin_amdgcn_wmma_i32_16x16x64_iu8(true, a0, true, b1, acc01, false, false);
            acc10 = __builtin_amdgcn_wmma_i32_16x16x64_iu8(true, a1, true, b0, acc10, false, false);
            acc11 = __builtin_amdgcn_wmma_i32_16x16x64_iu8(true, a1, true, b1, acc11, false, false);
        }
    }

    #pragma unroll
    for (int r = 0; r < 8; ++r) {
        const int mm0 = mbase + r + khalf * 8;
        const int mm1 = mm0 + 16;
        out[(size_t)mm0 * O + obase + row]      = (float)acc00[r];
        out[(size_t)mm0 * O + obase + 16 + row] = (float)acc01[r];
        out[(size_t)mm1 * O + obase + row]      = (float)acc10[r];
        out[(size_t)mm1 * O + obase + 16 + row] = (float)acc11[r];
    }
}

__global__ void bnstats_kernel(const float* __restrict__ y, float* __restrict__ s1,
                               float* __restrict__ s2, int M, int C, int parts) {
    int idx = blockIdx.x * blockDim.x + threadIdx.x;
    if (idx >= C * parts) return;
    int c    = idx % C;
    int part = idx / C;
    int chunk = M / parts;
    const float* p = y + (size_t)part * chunk * C + c;
    float a = 0.f, b = 0.f;
    for (int r = 0; r < chunk; ++r) {
        float v = p[(size_t)r * C];
        a += v;
        b += v * v;
    }
    atomicAdd(&s1[c], a);
    atomicAdd(&s2[c], b);
}

__global__ void bnfinalize_kernel(const float* __restrict__ s1, const float* __restrict__ s2,
                                  const float* __restrict__ g, const float* __restrict__ bt,
                                  float* __restrict__ scale, float* __restrict__ shift,
                                  int C, float invM) {
    int c = blockIdx.x * blockDim.x + threadIdx.x;
    if (c >= C) return;
    float mean = s1[c] * invM;
    float var  = s2[c] * invM - mean * mean;
    float sc   = g[c] * rsqrtf(var + 1e-5f);
    scale[c] = sc;
    shift[c] = bt[c] - mean * sc;
}

__global__ void bn_sign_kernel(const float* __restrict__ y, const float* __restrict__ scale,
                               const float* __restrict__ shift, signed char* __restrict__ out,
                               int M, int C) {
    int idx = blockIdx.x * blockDim.x + threadIdx.x;
    if (idx >= M * C) return;
    int c = idx % C;
    float v = y[idx] * scale[c] + shift[c];
    out[idx] = (v > 0.f) ? 1 : ((v < 0.f) ? -1 : 0);
}

__global__ void bn_pool_sign_kernel(const float* __restrict__ y, const float* __restrict__ scale,
                                    const float* __restrict__ shift, signed char* __restrict__ out,
                                    int N, int H, int W, int C) {
    int Ho = H >> 1, Wo = W >> 1;
    int idx = blockIdx.x * blockDim.x + threadIdx.x;
    if (idx >= N * Ho * Wo * C) return;
    int c  = idx % C;   int t = idx / C;
    int wo = t % Wo;    t /= Wo;
    int ho = t % Ho;    int n = t / Ho;
    float sc = scale[c], sh = shift[c];
    float m = -3.4e38f;
    #pragma unroll
    for (int dy = 0; dy < 2; ++dy)
        #pragma unroll
        for (int dx = 0; dx < 2; ++dx) {
            float v = y[((((size_t)n * H) + (2 * ho + dy)) * W + (2 * wo + dx)) * C + c] * sc + sh;
            m = fmaxf(m, v);
        }
    out[idx] = (m > 0.f) ? 1 : ((m < 0.f) ? -1 : 0);
}

__global__ void bn_pool_float_kernel(const float* __restrict__ y, const float* __restrict__ scale,
                                     const float* __restrict__ shift, float* __restrict__ out,
                                     int N, int H, int W, int C) {
    int Ho = H >> 1, Wo = W >> 1;
    int idx = blockIdx.x * blockDim.x + threadIdx.x;
    if (idx >= N * Ho * Wo * C) return;
    int c  = idx % C;   int t = idx / C;
    int wo = t % Wo;    t /= Wo;
    int ho = t % Ho;    int n = t / Ho;
    float sc = scale[c], sh = shift[c];
    float m = -3.4e38f;
    #pragma unroll
    for (int dy = 0; dy < 2; ++dy)
        #pragma unroll
        for (int dx = 0; dx < 2; ++dx) {
            float v = y[((((size_t)n * H) + (2 * ho + dy)) * W + (2 * wo + dx)) * C + c] * sc + sh;
            m = fmaxf(m, v);
        }
    out[idx] = m;
}

__global__ void fc_kernel(const float* __restrict__ p5, const float* __restrict__ fcw,
                          const float* __restrict__ fcb, float* __restrict__ out, int N) {
    int idx = blockIdx.x * blockDim.x + threadIdx.x;
    if (idx >= N * 10) return;
    int cls = idx % 10;
    int n   = idx / 10;
    float acc = fcb[cls];
    const float* wrow = fcw + (size_t)cls * 8192;
    for (int k = 0; k < 8192; ++k) {
        int c = k >> 4;
        int h = (k >> 2) & 3;
        int w = k & 3;
        acc += wrow[k] * p5[((((size_t)n * 4) + h) * 4 + w) * 512 + c];
    }
    out[idx] = acc;
}

static void bn_layer(hipStream_t s, const float* Y, int M, int C,
                     const float* g, const float* b,
                     float* S1, float* S2, float* SC, float* SH) {
    hipMemsetAsync(S1, 0, (size_t)C * sizeof(float), s);
    hipMemsetAsync(S2, 0, (size_t)C * sizeof(float), s);
    const int parts = 256;
    bnstats_kernel<<<(C * parts + 255) / 256, 256, 0, s>>>(Y, S1, S2, M, C, parts);
    bnfinalize_kernel<<<(C + 63) / 64, 64, 0, s>>>(S1, S2, g, b, SC, SH, C, 1.0f / (float)M);
}

static void binconv_layer(hipStream_t s, const signed char* in, const float* w,
                          float* out, signed char* wp,
                          int N, int H, int W, int C, int O) {
    const int K = 9 * C;
    packw_kernel<<<(O * K + 255) / 256, 256, 0, s>>>(w, wp, O, C);
    const int M = N * H * W;
    dim3 grid(M / 256, O / 32);
    binconv_wmma_kernel<<<grid, 256, (size_t)32 * K, s>>>(in, wp, out, N, H, W, C, O);
}

extern "C" void kernel_launch(void* const* d_in, const int* in_sizes, int n_in,
                              void* d_out, int out_size, void* d_ws, size_t ws_size,
                              hipStream_t stream) {
    const float* x   = (const float*)d_in[0];
    const float* w0  = (const float*)d_in[1];
    const float* wconv[5] = { (const float*)d_in[2], (const float*)d_in[3],
                              (const float*)d_in[4], (const float*)d_in[5],
                              (const float*)d_in[6] };
    const float* fcw = (const float*)d_in[7];
    const float* fcb = (const float*)d_in[8];
    const float* g[6]; const float* b[6];
    for (int i = 0; i < 6; ++i) {
        g[i] = (const float*)d_in[9 + 2 * i];
        b[i] = (const float*)d_in[10 + 2 * i];
    }

    char* ws = (char*)d_ws;
    float*       Y  = (float*)(ws + OFF_Y);
    signed char* B  = (signed char*)(ws + OFF_B);
    signed char* WP = (signed char*)(ws + OFF_WP);
    float*       S1 = (float*)(ws + OFF_S1);
    float*       S2 = (float*)(ws + OFF_S2);
    float*       SC = (float*)(ws + OFF_SC);
    float*       SH = (float*)(ws + OFF_SH);
    float*       P5 = (float*)(ws + OFF_P5);
    float*       out = (float*)d_out;

    const int N = 256;

    conv0_kernel<<<(N * 32 * 32 * 128) / 256, 256, 0, stream>>>(x, w0, Y, N, 32, 32, 3, 128);
    bn_layer(stream, Y, N * 32 * 32, 128, g[0], b[0], S1, S2, SC, SH);
    bn_sign_kernel<<<(N * 32 * 32 * 128 + 255) / 256, 256, 0, stream>>>(Y, SC, SH, B, N * 32 * 32, 128);

    binconv_layer(stream, B, wconv[0], Y, WP, N, 32, 32, 128, 128);
    bn_layer(stream, Y, N * 32 * 32, 128, g[1], b[1], S1, S2, SC, SH);
    bn_pool_sign_kernel<<<(N * 16 * 16 * 128 + 255) / 256, 256, 0, stream>>>(Y, SC, SH, B, N, 32, 32, 128);

    binconv_layer(stream, B, wconv[1], Y, WP, N, 16, 16, 128, 256);
    bn_layer(stream, Y, N * 16 * 16, 256, g[2], b[2], S1, S2, SC, SH);
    bn_sign_kernel<<<(N * 16 * 16 * 256 + 255) / 256, 256, 0, stream>>>(Y, SC, SH, B, N * 16 * 16, 256);

    binconv_layer(stream, B, wconv[2], Y, WP, N, 16, 16, 256, 256);
    bn_layer(stream, Y, N * 16 * 16, 256, g[3], b[3], S1, S2, SC, SH);
    bn_pool_sign_kernel<<<(N * 8 * 8 * 256 + 255) / 256, 256, 0, stream>>>(Y, SC, SH, B, N, 16, 16, 256);

    binconv_layer(stream, B, wconv[3], Y, WP, N, 8, 8, 256, 512);
    bn_layer(stream, Y, N * 8 * 8, 512, g[4], b[4], S1, S2, SC, SH);
    bn_sign_kernel<<<(N * 8 * 8 * 512 + 255) / 256, 256, 0, stream>>>(Y, SC, SH, B, N * 8 * 8, 512);

    binconv_layer(stream, B, wconv[4], Y, WP, N, 8, 8, 512, 512);
    bn_layer(stream, Y, N * 8 * 8, 512, g[5], b[5], S1, S2, SC, SH);
    bn_pool_float_kernel<<<(N * 4 * 4 * 512 + 255) / 256, 256, 0, stream>>>(Y, SC, SH, P5, N, 8, 8, 512);

    fc_kernel<<<(N * 10 + 255) / 256, 256, 0, stream>>>(P5, fcw, fcb, out, N);
}
